// Model_45466523795582
// MI455X (gfx1250) — matrix operand order repaired and checked by reading, not compiled, not run on hardware
//
#include <hip/hip_runtime.h>
#include <hip/hip_bf16.h>

// ---------------------------------------------------------------------------
// Rel-position MHA (Transformer-XL style), CDNA5 / gfx1250, wave32 WMMA bf16.
//   B=8, T=1024, F=512, H=8, DK=64, P=2*T-1=2047
// Pipeline:
//   1) cvt_kernel: one-shot f32 -> bf16 of activations + weights (packed cvt)
//   2) proj_kernel<MODE>: pure-bf16 WMMA GEMMs for q/k/v/p projections
//   3) attn_kernel: fused rel-shift attention (sliding bd window + softmax)
//   4) outproj_kernel: final bf16 WMMA GEMM -> f32 output
// Operand layouts: WMMA B fragment wants memory [n][k] row-major (lane = n):
//     ac   : K stored [B,H,T,DK]   (n=s, k=d)
//     bd   : P stored [H,P,DK]     (n=u, k=d)
//     attnV: V stored [B,H,DK,T]   (n=d, k=s)
// ---------------------------------------------------------------------------

typedef __attribute__((ext_vector_type(16))) __bf16 v16bf;
typedef __attribute__((ext_vector_type(4)))  __bf16 bf16x4;
typedef __attribute__((ext_vector_type(2)))  __bf16 bf16x2;
typedef __attribute__((ext_vector_type(8)))  float  v8f;

#define B_  8
#define T_  1024
#define F_  512
#define H_  8
#define DK_ 64
#define P_  2047

__device__ __forceinline__ __bf16 f2bf(float f)  { return (__bf16)f; }
__device__ __forceinline__ float  bf2f(__bf16 b) { return (float)b;  }
__device__ __forceinline__ bf16x2 ldbf2(const __bf16* p) {
  return *(const bf16x2*)p;
}

__device__ __forceinline__ v8f wmma_bf16(v16bf a, v16bf b, v8f c) {
  // (neg_a, A, neg_b, B, c_mod, C, reuse_a, reuse_b)
  return __builtin_amdgcn_wmma_f32_16x16x32_bf16(false, a, false, b, (short)0, c,
                                                 false, false);
}

// A-fragment (16-bit, 16x32): lane L: row M = L%16, half = L/16.
// here element pair v holds K0 = half*16 + 2v (contiguous k, not AMD's table order; B matches)
__device__ __forceinline__ int a_koff(int v, int half) {
  return (half << 4) + (v << 1);
}
// B-fragment (16-bit, 32x16): lane L: col N = L%16, half = L/16.
// here element pair v holds K0 = half*16 + 2v (same contiguous order as A)  -> 16 contiguous bf16 per lane.

__device__ __forceinline__ float wave_max32(float v) {
  #pragma unroll
  for (int off = 16; off; off >>= 1) v = fmaxf(v, __shfl_xor(v, off, 32));
  return v;
}
__device__ __forceinline__ float wave_sum32(float v) {
  #pragma unroll
  for (int off = 16; off; off >>= 1) v += __shfl_xor(v, off, 32);
  return v;
}

// ---------------------------------------------------------------------------
// One-shot f32 -> bf16 conversion (n must be a multiple of 4; all our tensors
// are). b128 load -> 2x v_cvt_pk_bf16_f32 -> b64 store.
// ---------------------------------------------------------------------------
__global__ void __launch_bounds__(256)
cvt_kernel(const float* __restrict__ in, __bf16* __restrict__ out, int n)
{
  const int i = (int)(blockIdx.x * 256 + threadIdx.x) << 2;
  if (i >= n) return;
  float4 v = *(const float4*)(in + i);
  bf16x4 r;
  r[0] = (__bf16)v.x; r[1] = (__bf16)v.y;
  r[2] = (__bf16)v.z; r[3] = (__bf16)v.w;
  *(bf16x4*)(out + i) = r;
}

// ---------------------------------------------------------------------------
// Projection: out[16x64 strip] = X[M,512] @ W[N=512,K=512]^T (+bias) -> bf16
//  MODE 0: q -> QU=q+pos_bias_u, QV=q+pos_bias_v, layout [B,H,T,DK]
//  MODE 1: v -> V^T layout [B,H,DK,T]
//  MODE 2: k -> K   layout [B,H,T,DK]
//  MODE 3: p -> P   layout [H,P,DK]   (no bias; M=2047: clamp + store guard)
// grid = (mtiles, 8 col strips), block = 32 (one wave). Pure bf16 operands.
// ---------------------------------------------------------------------------
template <int MODE>
__global__ void __launch_bounds__(32)
proj_kernel(const __bf16* __restrict__ Xb, const __bf16* __restrict__ Wb,
            const float* __restrict__ bias,
            const float* __restrict__ pbu, const float* __restrict__ pbv,
            __bf16* __restrict__ out0, __bf16* __restrict__ out1, int M)
{
  const int lane  = threadIdx.x;
  const int nlane = lane & 15;
  const int half  = lane >> 4;
  const int m0 = blockIdx.x << 4;
  const int n0 = blockIdx.y << 6;

  int arow = m0 + nlane;
  if (MODE == 3) arow = min(arow, M - 1);   // clamp: no divergent load guard
  const __bf16* xrow = Xb + (size_t)arow * F_;

  v8f acc[4] = {};
  for (int kb = 0; kb < F_; kb += 32) {
    v16bf a;
    #pragma unroll
    for (int v = 0; v < 8; ++v) {
      bf16x2 x2 = ldbf2(xrow + kb + a_koff(v, half));
      a[2*v] = x2[0]; a[2*v+1] = x2[1];
    }
    #pragma unroll
    for (int nt = 0; nt < 4; ++nt) {
      const __bf16* wrow =
          Wb + (size_t)(n0 + (nt << 4) + nlane) * F_ + kb + (half << 4);
      v16bf bb;
      #pragma unroll
      for (int v = 0; v < 8; ++v) {
        bf16x2 w2 = ldbf2(wrow + (v << 1));
        bb[2*v] = w2[0]; bb[2*v+1] = w2[1];
      }
      acc[nt] = wmma_bf16(a, bb, acc[nt]);
    }
  }

  #pragma unroll
  for (int nt = 0; nt < 4; ++nt) {
    const int n = n0 + (nt << 4) + nlane;
    const int h = n >> 6, d = n & 63;
    #pragma unroll
    for (int r = 0; r < 8; ++r) {
      const int m   = (half << 3) + r;
      const int row = m0 + m;
      if (MODE == 3 && row >= M) continue;
      float val = acc[nt][r];
      if (MODE != 3) val += bias[n];
      if (MODE == 0) {
        const int b = row >> 10, t = row & 1023;
        const size_t idx = (((size_t)b * H_ + h) * T_ + t) * DK_ + d;
        out0[idx] = f2bf(val + pbu[n]);
        out1[idx] = f2bf(val + pbv[n]);
      } else if (MODE == 1) {            // V^T [B,H,DK,T]
        const int b = row >> 10, t = row & 1023;
        out0[(((size_t)b * H_ + h) * DK_ + d) * T_ + t] = f2bf(val);
      } else if (MODE == 2) {            // K [B,H,T,DK]
        const int b = row >> 10, t = row & 1023;
        out0[(((size_t)b * H_ + h) * T_ + t) * DK_ + d] = f2bf(val);
      } else {                           // P [H,P,DK]
        out0[((size_t)h * P_ + row) * DK_ + d] = f2bf(val);
      }
    }
  }
}

// ---------------------------------------------------------------------------
// Attention: one workgroup (1 wave) per (b, h, 16-row t-tile).
// scores[t,s] = (qu[t]·k[s] + qv[t]·p[T-1-t+s]) / 8 ; softmax over s ; ×V.
// rel_shift fused: bd_g[t,u] = qv[t]·p[u] with u = s - t + T-1 is a plain
// GEMM in (t,u); the needed u-window slides by 16 per s-tile -> 16x32
// circular LDS window, one new WMMA bd tile per s-tile.
// NOTE: Pb must be padded by >= 1 row (128 B); the final window tile reads up
// to u = 2047 (one row past P), those slots are never consumed.
// ---------------------------------------------------------------------------
__global__ void __launch_bounds__(32)
attn_kernel(const __bf16* __restrict__ QU, const __bf16* __restrict__ QV,
            const __bf16* __restrict__ Kb, const __bf16* __restrict__ VT,
            const __bf16* __restrict__ Pb, const int* __restrict__ mask,
            __bf16* __restrict__ Xo)
{
  __shared__ __bf16 sc[16][T_];      // 32 KB: scores, then softmax probs (bf16)
  __shared__ float  bdbuf[16][32];   // 2 KB : circular window of bd_g tiles

  const int lane  = threadIdx.x;
  const int nlane = lane & 15;
  const int half  = lane >> 4;
  const int tt = blockIdx.x & 63;
  const int bh = blockIdx.x >> 6;
  const int b  = bh >> 3, h = bh & 7;
  const int t0 = tt << 4;
  const int ubase = T_ - 16 - t0;    // 16-aligned, >= 0

  // Preload A fragments for qu/qv rows t0..t0+15 (K = DK = 64 -> 2 frags each)
  v16bf aqu[2], aqv[2];
  const __bf16* qu_base = QU + ((size_t)bh * T_ + t0 + nlane) * DK_;
  const __bf16* qv_base = QV + ((size_t)bh * T_ + t0 + nlane) * DK_;
  #pragma unroll
  for (int ks = 0; ks < 2; ++ks) {
    #pragma unroll
    for (int v = 0; v < 8; ++v) {
      int k = ks * 32 + a_koff(v, half);
      bf16x2 pu = ldbf2(qu_base + k);
      bf16x2 pv = ldbf2(qv_base + k);
      aqu[ks][2*v] = pu[0]; aqu[ks][2*v+1] = pu[1];
      aqv[ks][2*v] = pv[0]; aqv[ks][2*v+1] = pv[1];
    }
  }

  // bd_g tile for u = ubase+uofs .. +15 into circular cols (uofs..uofs+15)&31
  const __bf16* p_head = Pb + (size_t)h * P_ * DK_;
  auto bd_tile = [&](int uofs) {
    const int u = ubase + uofs + nlane;
    const __bf16* pu = p_head + (size_t)u * DK_ + (half << 4);
    v8f c = {};
    #pragma unroll
    for (int ks = 0; ks < 2; ++ks) {
      v16bf bb;
      #pragma unroll
      for (int v = 0; v < 8; ++v) {
        bf16x2 pv = ldbf2(pu + ks * 32 + (v << 1));   // unguarded: ws padded
        bb[2*v] = pv[0]; bb[2*v+1] = pv[1];
      }
      c = wmma_bf16(aqv[ks], bb, c);
    }
    const int cc = (uofs + nlane) & 31;
    #pragma unroll
    for (int r = 0; r < 8; ++r) bdbuf[(half << 3) + r][cc] = c[r];
  };

  const __bf16* k_head = Kb + (size_t)bh * T_ * DK_;

  bd_tile(0);
  for (int st = 0; st < 64; ++st) {
    const int s0 = st << 4;
    __syncthreads();
    bd_tile(s0 + 16);                 // window now covers offsets s0..s0+31

    // ac tile: A = aqu, B = K[s0+n, d]
    const int s = s0 + nlane;
    const __bf16* kp = k_head + (size_t)s * DK_ + (half << 4);
    v8f acc = {};
    #pragma unroll
    for (int ks = 0; ks < 2; ++ks) {
      v16bf bb;
      #pragma unroll
      for (int v = 0; v < 8; ++v) {
        bf16x2 kv = ldbf2(kp + ks * 32 + (v << 1));
        bb[2*v] = kv[0]; bb[2*v+1] = kv[1];
      }
      acc = wmma_bf16(aqu[ks], bb, acc);
    }
    __syncthreads();

    const int mk = mask[(size_t)b * T_ + s];
    #pragma unroll
    for (int r = 0; r < 8; ++r) {
      const int m = (half << 3) + r;
      float bdv = bdbuf[m][(s - m + 15) & 31];   // bd_g[t, s - t + T-1]
      float v   = (acc[r] + bdv) * 0.125f;       // 1/sqrt(64)
      if (mk == 0) v = -1.0e30f;
      sc[m][s] = f2bf(v);
    }
  }
  __syncthreads();

  // Softmax per row (wave-parallel), write normalized probs back as bf16.
  for (int r = 0; r < 16; ++r) {
    float mx = -3.0e38f;
    for (int c = lane; c < T_; c += 32) mx = fmaxf(mx, bf2f(sc[r][c]));
    mx = wave_max32(mx);
    float sum = 0.f;
    for (int c = lane; c < T_; c += 32) sum += __expf(bf2f(sc[r][c]) - mx);
    sum = wave_sum32(sum);
    const float inv = 1.0f / sum;
    for (int c = lane; c < T_; c += 32)
      sc[r][c] = f2bf(__expf(bf2f(sc[r][c]) - mx) * inv);
  }
  __syncthreads();

  // out[16, 64] = probs[16,1024] @ V[1024,64], V stored transposed [DK,T].
  v8f o[4] = {};
  for (int kb = 0; kb < T_; kb += 32) {
    v16bf a;
    #pragma unroll
    for (int v = 0; v < 8; ++v) {
      int k = kb + a_koff(v, half);
      bf16x2 pr = *(const bf16x2*)(&sc[nlane][k]);
      a[2*v] = pr[0]; a[2*v+1] = pr[1];
    }
    #pragma unroll
    for (int nt = 0; nt < 4; ++nt) {
      const int d = (nt << 4) + nlane;
      const __bf16* vp = VT + ((size_t)bh * DK_ + d) * T_ + kb + (half << 4);
      v16bf bb;
      #pragma unroll
      for (int v = 0; v < 8; ++v) {
        bf16x2 vv = ldbf2(vp + (v << 1));
        bb[2*v] = vv[0]; bb[2*v+1] = vv[1];
      }
      o[nt] = wmma_bf16(a, bb, o[nt]);
    }
  }

  #pragma unroll
  for (int nt = 0; nt < 4; ++nt) {
    #pragma unroll
    for (int r = 0; r < 8; ++r) {
      const int m = (half << 3) + r;
      const size_t idx = ((size_t)b * T_ + t0 + m) * F_ + h * DK_ + (nt << 4) + nlane;
      Xo[idx] = f2bf(o[nt][r]);
    }
  }
}

// ---------------------------------------------------------------------------
// Output projection: out[8192,512] (f32) = Xo(bf16) @ Wo^T(bf16) + bo
// ---------------------------------------------------------------------------
__global__ void __launch_bounds__(32)
outproj_kernel(const __bf16* __restrict__ Xb, const __bf16* __restrict__ Wb,
               const float* __restrict__ bias, float* __restrict__ out)
{
  const int lane  = threadIdx.x;
  const int nlane = lane & 15;
  const int half  = lane >> 4;
  const int m0 = blockIdx.x << 4;
  const int n0 = blockIdx.y << 6;

  v8f acc[4] = {};
  const __bf16* xrow = Xb + (size_t)(m0 + nlane) * F_;

  for (int kb = 0; kb < F_; kb += 32) {
    v16bf a;
    #pragma unroll
    for (int v = 0; v < 8; ++v) {
      bf16x2 xv = ldbf2(xrow + kb + a_koff(v, half));
      a[2*v] = xv[0]; a[2*v+1] = xv[1];
    }
    #pragma unroll
    for (int nt = 0; nt < 4; ++nt) {
      const __bf16* wrow =
          Wb + (size_t)(n0 + (nt << 4) + nlane) * F_ + kb + (half << 4);
      v16bf bb;
      #pragma unroll
      for (int v = 0; v < 8; ++v) {
        bf16x2 wv = ldbf2(wrow + (v << 1));
        bb[2*v] = wv[0]; bb[2*v+1] = wv[1];
      }
      acc[nt] = wmma_bf16(a, bb, acc[nt]);
    }
  }

  #pragma unroll
  for (int nt = 0; nt < 4; ++nt) {
    const int n = n0 + (nt << 4) + nlane;
    #pragma unroll
    for (int r = 0; r < 8; ++r) {
      const int row = m0 + (half << 3) + r;
      out[(size_t)row * F_ + n] = acc[nt][r] + bias[n];
    }
  }
}

// ---------------------------------------------------------------------------
extern "C" void kernel_launch(void* const* d_in, const int* in_sizes, int n_in,
                              void* d_out, int out_size, void* d_ws, size_t ws_size,
                              hipStream_t stream)
{
  const float* query  = (const float*)d_in[0];
  const float* key_in = (const float*)d_in[1];
  const float* value  = (const float*)d_in[2];
  const float* pos    = (const float*)d_in[3];
  const int*   mask   = (const int*)  d_in[4];
  const float* Wq  = (const float*)d_in[5];
  const float* bq  = (const float*)d_in[6];
  const float* Wk  = (const float*)d_in[7];
  const float* bk  = (const float*)d_in[8];
  const float* Wv  = (const float*)d_in[9];
  const float* bv  = (const float*)d_in[10];
  const float* Wo  = (const float*)d_in[11];
  const float* bo  = (const float*)d_in[12];
  const float* Wp  = (const float*)d_in[13];
  const float* pbu = (const float*)d_in[14];
  const float* pbv = (const float*)d_in[15];

  // Workspace carve-up (bf16 elements).
  __bf16* ws = (__bf16*)d_ws;
  size_t off = 0;
  auto alloc = [&](size_t n) { __bf16* p = ws + off; off += n; return p; };

  const size_t NACT = (size_t)B_ * T_ * F_;        // 4,194,304
  const size_t NPOS = (size_t)P_ * F_;             // 1,048,064
  const size_t NW   = (size_t)F_ * F_;             //   262,144

  __bf16* Qx  = alloc(NACT);                       // bf16 activations
  __bf16* Kx  = alloc(NACT);
  __bf16* Vx  = alloc(NACT);
  __bf16* Px  = alloc(NPOS + 512);
  __bf16* Wqb = alloc(NW);                         // bf16 weights
  __bf16* Wkb = alloc(NW);
  __bf16* Wvb = alloc(NW);
  __bf16* Wpb = alloc(NW);
  __bf16* Wob = alloc(NW);
  __bf16* QU  = alloc(NACT);                       // projected operands
  __bf16* QV  = alloc(NACT);
  __bf16* Kb  = alloc(NACT);
  __bf16* VT  = alloc(NACT);
  __bf16* Pb  = alloc((size_t)H_ * P_ * DK_ + 1024);  // +pad: bd over-read
  __bf16* Xo  = alloc(NACT);                       // attention output

  dim3 blk(32);
  auto cvt = [&](const float* in, __bf16* out, size_t n) {
    cvt_kernel<<<dim3((unsigned)((n / 4 + 255) / 256)), dim3(256), 0, stream>>>(
        in, out, (int)n);
  };
  cvt(query,  Qx, NACT);
  cvt(key_in, Kx, NACT);
  cvt(value,  Vx, NACT);
  cvt(pos,    Px, NPOS);
  cvt(Wq, Wqb, NW);
  cvt(Wk, Wkb, NW);
  cvt(Wv, Wvb, NW);
  cvt(Wp, Wpb, NW);
  cvt(Wo, Wob, NW);

  proj_kernel<0><<<dim3(512, 8), blk, 0, stream>>>(Qx, Wqb, bq, pbu, pbv,
                                                   QU, QV, B_ * T_);
  proj_kernel<2><<<dim3(512, 8), blk, 0, stream>>>(Kx, Wkb, bk, nullptr, nullptr,
                                                   Kb, nullptr, B_ * T_);
  proj_kernel<1><<<dim3(512, 8), blk, 0, stream>>>(Vx, Wvb, bv, nullptr, nullptr,
                                                   VT, nullptr, B_ * T_);
  proj_kernel<3><<<dim3(128, 8), blk, 0, stream>>>(Px, Wpb, nullptr, nullptr, nullptr,
                                                   Pb, nullptr, P_);
  attn_kernel<<<dim3(B_ * H_ * (T_ / 16)), blk, 0, stream>>>(QU, QV, Kb, VT, Pb,
                                                             mask, Xo);
  outproj_kernel<<<dim3(512, 8), blk, 0, stream>>>(Xo, Wob, bo, (float*)d_out);
}
